// UGRNN_14413910245615
// MI455X (gfx1250) — matrix operand order repaired and checked by reading, not compiled, not run on hardware
//
#include <hip/hip_runtime.h>
#include <cmath>

// ---------------------------------------------------------------------------
// UGRNN scan for MI455X (gfx1250): persistent kernel, bf16 WMMA, LDS-staged
// weights, device-wide barrier once per timestep, double-buffered th state.
// K-loops software-pipelined 4-deep so WMMAs overlap global/LDS loads.
// ---------------------------------------------------------------------------

typedef __attribute__((ext_vector_type(16))) __bf16 v16bf;
typedef __attribute__((ext_vector_type(8)))  float  v8f;

#define S_DIM   1024
#define O_DIM   128
#define B_DIM   64
#define T_STEPS 512
#define NWG     64
#define LDS_K   (S_DIM + 8)   // padded K-row (bf16) -> 2064B stride, conflict-free b128 reads

union ABVec { v16bf v; uint4 q[2]; };

__device__ __forceinline__ v8f wmma_bf16(v16bf a, v16bf b, v8f c) {
  return __builtin_amdgcn_wmma_f32_16x16x32_bf16(false, a, false, b, (short)0, c, false, false);
}

// Branch-free, overflow-safe tanh: 1 - 2/(exp(2x)+1)  (inf -> +1, 0 -> -1 limits ok)
__device__ __forceinline__ float fast_tanh(float v) {
  float e = __expf(v + v);
  return 1.0f - 2.0f / (e + 1.0f);
}
__device__ __forceinline__ float fast_sigmoid(float v) {
  return 1.0f / (1.0f + __expf(-v));
}

__device__ __forceinline__ void grid_barrier(unsigned* cnt, unsigned epoch) {
  __threadfence();                 // release this wave's global stores
  __syncthreads();
  if (threadIdx.x == 0) {
    __hip_atomic_fetch_add(cnt, 1u, __ATOMIC_RELEASE, __HIP_MEMORY_SCOPE_AGENT);
    while (__hip_atomic_load(cnt, __ATOMIC_ACQUIRE, __HIP_MEMORY_SCOPE_AGENT) < epoch * NWG) {
      __builtin_amdgcn_s_sleep(2);
    }
  }
  __syncthreads();
  __threadfence();                 // acquire: invalidate stale WGP$ lines for all waves
}

// Reset barrier counter and seed th0 = tanh(h_init) in bf16 (deterministic per launch).
__global__ void ugrnn_init(const float* __restrict__ h_init,
                           __bf16* __restrict__ thbuf,
                           unsigned* __restrict__ cnt) {
  int i = blockIdx.x * blockDim.x + threadIdx.x;
  if (i == 0) *cnt = 0u;
  for (int idx = i; idx < B_DIM * S_DIM; idx += (int)(gridDim.x * blockDim.x))
    thbuf[idx] = (__bf16)fast_tanh(h_init[idx]);
}

__global__ __launch_bounds__(256, 1)
void ugrnn_persistent(const float* __restrict__ x,
                      const float* __restrict__ h_init,
                      const float* __restrict__ w_o,
                      const float* __restrict__ b_o,
                      const float* __restrict__ w_u,
                      const float* __restrict__ b_u,
                      const float* __restrict__ w_r,
                      const float* __restrict__ b_r,
                      float* __restrict__ out,
                      __bf16* __restrict__ thbuf,
                      unsigned* __restrict__ cnt) {
  extern __shared__ char smem[];
  __bf16* wuT = (__bf16*)smem;                 // [16][LDS_K]  w_u slice, transposed (n,k)
  __bf16* wrT = wuT + 16 * LDS_K;              // [16][LDS_K]  w_r slice
  __bf16* woT = wrT + 16 * LDS_K;              // [16][LDS_K]  w_o slice (WGs 0..31)
  float*  hls = (float*)(woT + 16 * LDS_K);    // [64][16]     h state slice (f32)
  float*  xch = hls + B_DIM * 16;              // [8][256]     cross-wave exchange

  const int wg   = blockIdx.x;
  const int tid  = threadIdx.x;
  const int lane = tid & 31;
  const int wave = tid >> 5;
  const int hh   = lane >> 4;                  // lane half (K-group select)
  const int ln   = lane & 15;                  // M-row (A) / N-col (B)
  const int col0 = wg * 16;                    // this WG's state-column slice
  const int mt   = wave & 3;                   // M-tile (batch rows 16*mt..)
  const int msel = wave >> 2;                  // 0: u-gate GEMM, 1: candidate GEMM
  const int mt3  = wg >> 3;                    // output-GEMM tile coords (WGs 0..31)
  const int nt3  = wg & 7;

  // ---- one-time: stage bf16 weight slices into LDS, load h slice ----
  for (int idx = tid; idx < 16 * S_DIM; idx += 256) {
    int n = idx >> 10, k = idx & (S_DIM - 1);
    wuT[n * LDS_K + k] = (__bf16)w_u[k * S_DIM + col0 + n];
    wrT[n * LDS_K + k] = (__bf16)w_r[k * S_DIM + col0 + n];
    if (wg < 32) woT[n * LDS_K + k] = (__bf16)w_o[(nt3 * 16 + n) * S_DIM + k];
  }
  for (int idx = tid; idx < B_DIM * 16; idx += 256)
    hls[idx] = h_init[(idx >> 4) * S_DIM + col0 + (idx & 15)];
  const float bu = b_u[col0 + ln];
  const float br = b_r[col0 + ln];
  __syncthreads();

  #pragma unroll 1
  for (int t = 0; t < T_STEPS; ++t) {
    const __bf16* thcur  = thbuf + (t & 1) * (B_DIM * S_DIM);
    __bf16*       thnext = thbuf + ((t + 1) & 1) * (B_DIM * S_DIM);

    // ---- phase 1: h_new for this WG's 16 columns (two 64x16x1024 GEMMs) ----
    v8f acc = {};
    {
      const __bf16* abase = thcur + (mt * 16 + ln) * S_DIM + 8 * hh;      // A: th rows
      const __bf16* bbase = (msel ? wrT : wuT) + ln * LDS_K + 8 * hh;     // B: LDS weights
      auto LD = [&](ABVec& a, ABVec& b, int k0) {
        a.q[0] = *(const uint4*)(abase + k0);        // K = k0+8h   .. +7   (A layout)
        a.q[1] = *(const uint4*)(abase + k0 + 16);   // K = k0+16+8h.. +7
        b.q[0] = *(const uint4*)(bbase + k0);        // K = k0+8h   .. +7   (B layout)
        b.q[1] = *(const uint4*)(bbase + k0 + 16);   // K = k0+16+8h.. +7
      };
      ABVec a0, b0, a1, b1, a2, b2, a3, b3;
      LD(a0, b0, 0);  LD(a1, b1, 32);  LD(a2, b2, 64);  LD(a3, b3, 96);
      #pragma unroll 1
      for (int k0 = 128; k0 < S_DIM; k0 += 128) {    // 4-deep rotating pipeline
        acc = wmma_bf16(a0.v, b0.v, acc);  LD(a0, b0, k0);
        acc = wmma_bf16(a1.v, b1.v, acc);  LD(a1, b1, k0 + 32);
        acc = wmma_bf16(a2.v, b2.v, acc);  LD(a2, b2, k0 + 64);
        acc = wmma_bf16(a3.v, b3.v, acc);  LD(a3, b3, k0 + 96);
      }
      acc = wmma_bf16(a0.v, b0.v, acc);
      acc = wmma_bf16(a1.v, b1.v, acc);
      acc = wmma_bf16(a2.v, b2.v, acc);
      acc = wmma_bf16(a3.v, b3.v, acc);
    }
    if (msel == 1) {                                  // candidate waves publish
      #pragma unroll
      for (int r = 0; r < 8; ++r) xch[mt * 256 + r * 32 + lane] = acc[r];
    }
    __syncthreads();
    if (msel == 0) {                                  // gate waves fuse the update
      #pragma unroll
      for (int r = 0; r < 8; ++r) {
        int row = mt * 16 + r + 8 * hh;               // C/D layout: M = r + 8*half
        float ua = acc[r] + bu;
        float ra = xch[mt * 256 + r * 32 + lane] + br;
        float u  = fast_sigmoid(ua);
        float cd = fast_tanh(ra);
        float hp = hls[row * 16 + ln];
        float hn = u * hp + (1.0f - u) * cd;
        hls[row * 16 + ln] = hn;
        thnext[row * S_DIM + col0 + ln] = (__bf16)fast_tanh(hn);  // next step's A matrix
      }
    }

    grid_barrier(cnt, (unsigned)(t + 1));

    // ---- phase 2 (WGs 0..31): errors[t] 16x16 tile = th_{t+1} @ w_o^T + b_o - x[t] ----
    if (wg < 32) {
      int r0   = tid >> 5;
      int l2   = tid & 31;
      int orow = mt3 * 16 + (r0 + 8 * (l2 >> 4));
      int ocol = nt3 * 16 + (l2 & 15);
      size_t oidx = (size_t)t * (B_DIM * O_DIM) + (size_t)orow * O_DIM + ocol;
      __builtin_prefetch(&x[oidx], 0, 0);             // global_prefetch_b8

      v8f acc3 = {};
      const __bf16* abase = thnext + (mt3 * 16 + ln) * S_DIM + 8 * hh + wave * 128;
      const __bf16* bbase = woT + ln * LDS_K + 8 * hh + wave * 128;
      auto LD3 = [&](ABVec& a, ABVec& b, int k0) {
        a.q[0] = *(const uint4*)(abase + k0);
        a.q[1] = *(const uint4*)(abase + k0 + 16);
        b.q[0] = *(const uint4*)(bbase + k0);
        b.q[1] = *(const uint4*)(bbase + k0 + 16);
      };
      ABVec a0, b0, a1, b1, a2, b2, a3, b3;           // split-K: 128 per wave, all in flight
      LD3(a0, b0, 0);  LD3(a1, b1, 32);  LD3(a2, b2, 64);  LD3(a3, b3, 96);
      acc3 = wmma_bf16(a0.v, b0.v, acc3);
      acc3 = wmma_bf16(a1.v, b1.v, acc3);
      acc3 = wmma_bf16(a2.v, b2.v, acc3);
      acc3 = wmma_bf16(a3.v, b3.v, acc3);

      #pragma unroll
      for (int r = 0; r < 8; ++r) xch[wave * 256 + r * 32 + lane] = acc3[r];
      __syncthreads();
      {
        float s = 0.0f;
        #pragma unroll
        for (int w = 0; w < 8; ++w) s += xch[w * 256 + tid];
        out[oidx] = s + b_o[ocol] - x[oidx];
      }
      __syncthreads();                                 // protect xch before next phase-1
    }
  }
}

extern "C" void kernel_launch(void* const* d_in, const int* in_sizes, int n_in,
                              void* d_out, int out_size, void* d_ws, size_t ws_size,
                              hipStream_t stream) {
  (void)in_sizes; (void)n_in; (void)out_size; (void)ws_size;
  const float* x      = (const float*)d_in[0];
  const float* h_init = (const float*)d_in[1];
  const float* w_o    = (const float*)d_in[2];
  const float* b_o    = (const float*)d_in[3];
  const float* w_u    = (const float*)d_in[4];
  const float* b_u    = (const float*)d_in[5];
  const float* w_r    = (const float*)d_in[6];
  const float* b_r    = (const float*)d_in[7];
  float* out = (float*)d_out;

  unsigned* cnt  = (unsigned*)d_ws;                       // barrier counter
  __bf16* thbuf  = (__bf16*)((char*)d_ws + 128);          // th double buffer: 2*64*1024 bf16

  ugrnn_init<<<64, 256, 0, stream>>>(h_init, thbuf, cnt);

  size_t smem = (size_t)(3 * 16 * LDS_K) * sizeof(__bf16)       // wuT/wrT/woT
              + (size_t)(B_DIM * 16 + 8 * 256) * sizeof(float); // hls + xch  (~111 KB)
  ugrnn_persistent<<<NWG, 256, smem, stream>>>(x, h_init, w_o, b_o, w_u, b_u,
                                               w_r, b_r, out, thbuf, cnt);
}
